// TemporalDilatedAttention_85736137162902
// MI455X (gfx1250) — matrix operand order repaired and checked by reading, not compiled, not run on hardware
//
#include <hip/hip_runtime.h>

// ---------- types ----------
typedef __attribute__((ext_vector_type(16))) __bf16        v16bf;
typedef __attribute__((ext_vector_type(8)))  float         v8f;
typedef __attribute__((ext_vector_type(4))) unsigned int   v4u;
typedef __attribute__((ext_vector_type(4)))  float         v4f;

union Frag {                    // one WMMA A or B operand (16 bf16 per lane)
    v16bf bf;
    v4u   u4[2];
};

__device__ __forceinline__ unsigned short f2bf(float f) {
    unsigned int u = __float_as_uint(f);
    return (unsigned short)((u + 0x7FFFu + ((u >> 16) & 1u)) >> 16);  // RNE
}

#define WMMA_BF16(A, B, C) \
    __builtin_amdgcn_wmma_f32_16x16x32_bf16(false, (A), false, (B), (short)0, (C), false, false)

// ---------- CDNA5 async global->LDS path (guarded; fallback = sync copy) ----
#if __has_builtin(__builtin_amdgcn_global_load_async_to_lds_b128)
#define HAVE_ASYNC_LDS 1
#else
#define HAVE_ASYNC_LDS 0
#endif

// exact pointee type the builtin expects: 16-byte int vector
typedef int v4i_t __attribute__((vector_size(16)));
#define ASYNC_SRC(p) ((__attribute__((address_space(1))) v4i_t*)(p))
#define ASYNC_DST(p) ((__attribute__((address_space(3))) v4i_t*)(p))

__device__ __forceinline__ void wait_asynccnt0() {
#if __has_builtin(__builtin_amdgcn_s_wait_asynccnt)
    __builtin_amdgcn_s_wait_asynccnt(0);
#else
    asm volatile("s_wait_asynccnt 0x0" ::: "memory");
#endif
}

// =====================================================================
// GEMM:  out[M x N] = A[M x K](f32->bf16) @ W[K x N](f32->bf16) + bias
//        optional += residual ; out is bf16 (ushort) or f32
// Tiles: BM=64 BN=64 BK=32 ; 128 threads = 4 waves, each wave 32x32.
// =====================================================================
#define BM 64
#define BN 64
#define BK 32

template <bool OUT_BF16, bool RESID>
__global__ __launch_bounds__(128) void gemm_bias_kernel(
    const float* __restrict__ A, const float* __restrict__ W,
    const float* __restrict__ bias, const float* __restrict__ resid,
    void* __restrict__ out, int M, int N, int K)
{
    __shared__ unsigned short As[BM][BK + 8];   // [m][k], stride 40 (80B, 16B aligned)
    __shared__ unsigned short Bs[BN][BK + 8];   // [n][k] (W transposed)

    const int tid  = threadIdx.x;
    const int lane = tid & 31;
    const int wave = tid >> 5;
    const int wm   = wave >> 1;          // 0..1
    const int wn   = wave & 1;           // 0..1
    const int row0 = blockIdx.x * BM;
    const int col0 = blockIdx.y * BN;

    v8f zero = {};
    v8f acc[2][2];
    acc[0][0] = zero; acc[0][1] = zero; acc[1][0] = zero; acc[1][1] = zero;

    for (int k0 = 0; k0 < K; k0 += BK) {
        // --- stage A tile: thread -> row tid>>1, 16 cols ---
        {
            const int r  = tid >> 1;
            const int cb = (tid & 1) * 16;
            const float* src = A + (size_t)(row0 + r) * K + k0 + cb;
            #pragma unroll
            for (int j = 0; j < 16; j += 4) {
                v4f f = *(const v4f*)(src + j);
                As[r][cb + j + 0] = f2bf(f[0]);
                As[r][cb + j + 1] = f2bf(f[1]);
                As[r][cb + j + 2] = f2bf(f[2]);
                As[r][cb + j + 3] = f2bf(f[3]);
            }
        }
        // --- stage B tile transposed: thread -> k-row tid>>2, 16 cols ---
        {
            const int kr = tid >> 2;
            const int nb = (tid & 3) * 16;
            const float* src = W + (size_t)(k0 + kr) * N + col0 + nb;
            #pragma unroll
            for (int j = 0; j < 16; j += 4) {
                v4f f = *(const v4f*)(src + j);
                Bs[nb + j + 0][kr] = f2bf(f[0]);
                Bs[nb + j + 1][kr] = f2bf(f[1]);
                Bs[nb + j + 2][kr] = f2bf(f[2]);
                Bs[nb + j + 3][kr] = f2bf(f[3]);
            }
        }
        __syncthreads();

        // --- fragments + WMMA ---
        Frag af[2], bf[2];
        {
            const int rr = (lane & 15);
            const int kb = (lane >> 4) * 8;
            #pragma unroll
            for (int ms = 0; ms < 2; ms++) {
                const int row = wm * 32 + ms * 16 + rr;
                af[ms].u4[0] = *(const v4u*)&As[row][kb];
                af[ms].u4[1] = *(const v4u*)&As[row][kb + 16];
            }
            const int kbB = (lane >> 4) * 8;
            #pragma unroll
            for (int ns = 0; ns < 2; ns++) {
                const int col = wn * 32 + ns * 16 + rr;
                bf[ns].u4[0] = *(const v4u*)&Bs[col][kbB];
                bf[ns].u4[1] = *(const v4u*)&Bs[col][kbB + 16];
            }
        }
        #pragma unroll
        for (int ms = 0; ms < 2; ms++)
            #pragma unroll
            for (int ns = 0; ns < 2; ns++)
                acc[ms][ns] = WMMA_BF16(af[ms].bf, bf[ns].bf, acc[ms][ns]);
        __syncthreads();
    }

    // --- epilogue (C layout: lane L -> N = L%16, M = 8*(L/16)+r) ---
    const int g  = lane >> 4;
    const int nn = lane & 15;
    #pragma unroll
    for (int ms = 0; ms < 2; ms++) {
        #pragma unroll
        for (int ns = 0; ns < 2; ns++) {
            const int col = col0 + wn * 32 + ns * 16 + nn;
            const float bv = bias[col];
            #pragma unroll
            for (int r = 0; r < 8; r++) {
                const int row = row0 + wm * 32 + ms * 16 + g * 8 + r;
                const size_t idx = (size_t)row * N + col;
                float v = acc[ms][ns][r] + bv;
                if (RESID) v += resid[idx];
                if (OUT_BF16) ((unsigned short*)out)[idx] = f2bf(v);
                else          ((float*)out)[idx] = v;
            }
        }
    }
}

// =====================================================================
// Flash attention over dilated history.
// grid = (4 q-tiles of 64, H=4, 4 dilations), block = 128 (4 waves).
// Each wave: 16 query rows; Q frags in regs; stream 32-key blocks.
// K tile staged via GLOBAL_LOAD_ASYNC_TO_LDS_B128 (ASYNCcnt) when the
// toolchain exposes it; V tile is transposed on the fly (sync path).
// =====================================================================
__global__ __launch_bounds__(128) void attn_kernel(
    const unsigned short* __restrict__ Qb,   // [256][1024] bf16
    const unsigned short* __restrict__ Kb,   // [16384][1024] bf16 (row = l*256+b)
    const unsigned short* __restrict__ Vb,   // [16384][1024] bf16
    float* __restrict__ att)                 // [4][256][1024] f32
{
    __shared__ unsigned short Kt[32][264];   // [key][hd]  stride 264 (528B)
    __shared__ unsigned short Vt[256][40];   // [hd][key]  stride 40  (80B)
    __shared__ unsigned short Pl[4][16][40]; // per-wave P tile [q][key]

    const int dils[4] = {1, 4, 8, 16};
    const int m0 = blockIdx.x * 64;
    const int h  = blockIdx.y;
    const int di = blockIdx.z;
    const int d  = dils[di];
    const int T  = (64 / d) * 256;

    const int tid  = threadIdx.x;
    const int lane = tid & 31;
    const int wave = tid >> 5;
    const int qrow0 = m0 + wave * 16;
    const float scale = 0.0625f;            // 1/sqrt(256)

    // Q fragments (A layout, straight from global: row-major, contiguous k-chunks)
    Frag qf[8];
    {
        const int r  = qrow0 + (lane & 15);
        const int kb = (lane >> 4) * 8;
        const unsigned short* base = Qb + (size_t)r * 1024 + h * 256 + kb;
        #pragma unroll
        for (int kc = 0; kc < 8; kc++) {
            qf[kc].u4[0] = *(const v4u*)(base + kc * 32);
            qf[kc].u4[1] = *(const v4u*)(base + kc * 32 + 16);
        }
    }

    float mrun[8], lrun[8];
    #pragma unroll
    for (int r = 0; r < 8; r++) { mrun[r] = -1e30f; lrun[r] = 0.0f; }
    v8f zero = {};
    v8f oacc[16];
    #pragma unroll
    for (int s = 0; s < 16; s++) oacc[s] = zero;

    for (int t0 = 0; t0 < T; t0 += 32) {
        const int fi = t0 >> 8;                          // frame slot in strided set
        const int rowbase = fi * d * 256 + (t0 & 255);   // row in Kall/Vall

        const int key = tid >> 2;
        const int cb  = (tid & 3) * 64;
        const unsigned short* ksrc = Kb + (size_t)(rowbase + key) * 1024 + h * 256 + cb;
        const unsigned short* vsrc = Vb + (size_t)(rowbase + key) * 1024 + h * 256 + cb;

        // --- K tile: async DMA straight into LDS (per-lane gather, ASYNCcnt) ---
#if HAVE_ASYNC_LDS
        {
            unsigned short* kdst = &Kt[key][cb];
            #pragma unroll
            for (int j = 0; j < 64; j += 8)
                __builtin_amdgcn_global_load_async_to_lds_b128(
                    ASYNC_SRC(ksrc + j), ASYNC_DST(kdst + j), 0, 0);
        }
#else
        {
            #pragma unroll
            for (int j = 0; j < 64; j += 8)
                *(v4u*)&Kt[key][cb + j] = *(const v4u*)(ksrc + j);
        }
#endif
        // --- V tile: transpose during staging (sync path) ---
        {
            #pragma unroll
            for (int j = 0; j < 64; j += 8) {
                union { v4u v; unsigned short s[8]; } tmp;
                tmp.v = *(const v4u*)(vsrc + j);
                #pragma unroll
                for (int e = 0; e < 8; e++) Vt[cb + j + e][key] = tmp.s[e];
            }
        }
        // --- prefetch next key block while we compute on this one ---
        {
            const int t1 = t0 + 32;
            if (t1 < T) {
                const int rb1 = (t1 >> 8) * d * 256 + (t1 & 255);
                const unsigned short* kn = Kb + (size_t)(rb1 + key) * 1024 + h * 256 + cb;
                const unsigned short* vn = Vb + (size_t)(rb1 + key) * 1024 + h * 256 + cb;
                __builtin_prefetch(kn, 0, 0);
                __builtin_prefetch(vn, 0, 0);
            }
        }
#if HAVE_ASYNC_LDS
        wait_asynccnt0();
#endif
        __syncthreads();

        // --- scores: S = Q K^T (two 16x16 tiles = 32 keys) ---
        v8f s0 = zero, s1 = zero;
        {
            const int n  = lane & 15;
            const int kg = (lane >> 4) * 8;
            #pragma unroll
            for (int kc = 0; kc < 8; kc++) {
                Frag b0, b1;
                const int kb = kc * 32 + kg;
                b0.u4[0] = *(const v4u*)&Kt[n][kb];
                b0.u4[1] = *(const v4u*)&Kt[n][kb + 16];
                b1.u4[0] = *(const v4u*)&Kt[n + 16][kb];
                b1.u4[1] = *(const v4u*)&Kt[n + 16][kb + 16];
                s0 = WMMA_BF16(qf[kc].bf, b0.bf, s0);
                s1 = WMMA_BF16(qf[kc].bf, b1.bf, s1);
            }
        }

        // --- streaming softmax update (row m = 8*(lane/16)+r lives in a 16-lane group) ---
        float p0[8], p1[8];
        #pragma unroll
        for (int r = 0; r < 8; r++) {
            float a = s0[r] * scale, b = s1[r] * scale;
            float mx = fmaxf(a, b);
            mx = fmaxf(mx, __shfl_xor(mx, 1));
            mx = fmaxf(mx, __shfl_xor(mx, 2));
            mx = fmaxf(mx, __shfl_xor(mx, 4));
            mx = fmaxf(mx, __shfl_xor(mx, 8));
            const float mnew  = fmaxf(mrun[r], mx);
            const float alpha = __expf(mrun[r] - mnew);
            const float e0 = __expf(a - mnew);
            const float e1 = __expf(b - mnew);
            float rs = e0 + e1;
            rs += __shfl_xor(rs, 1);
            rs += __shfl_xor(rs, 2);
            rs += __shfl_xor(rs, 4);
            rs += __shfl_xor(rs, 8);
            lrun[r] = lrun[r] * alpha + rs;
            mrun[r] = mnew;
            p0[r] = e0; p1[r] = e1;
            #pragma unroll
            for (int s = 0; s < 16; s++) oacc[s][r] *= alpha;
        }

        // --- re-layout P via per-wave LDS patch (C layout -> A layout) ---
        {
            const int g = lane >> 4, n = lane & 15;
            #pragma unroll
            for (int r = 0; r < 8; r++) {
                Pl[wave][g * 8 + r][n]      = f2bf(p0[r]);
                Pl[wave][g * 8 + r][n + 16] = f2bf(p1[r]);
            }
        }
        Frag pf;
        {
            const int q  = lane & 15;
            const int kb = (lane >> 4) * 8;
            pf.u4[0] = *(const v4u*)&Pl[wave][q][kb];
            pf.u4[1] = *(const v4u*)&Pl[wave][q][kb + 16];
        }

        // --- O += P V  (16 hd-subtiles, K=32 keys per WMMA) ---
        {
            const int n  = lane & 15;
            const int kg = (lane >> 4) * 8;
            #pragma unroll
            for (int s = 0; s < 16; s++) {
                Frag vb;
                vb.u4[0] = *(const v4u*)&Vt[s * 16 + n][kg];
                vb.u4[1] = *(const v4u*)&Vt[s * 16 + n][kg + 16];
                oacc[s] = WMMA_BF16(pf.bf, vb.bf, oacc[s]);
            }
        }
        __syncthreads();
    }

    // --- finalize: O / l -> att[di] ---
    {
        const int g = lane >> 4, n = lane & 15;
        float inv[8];
        #pragma unroll
        for (int r = 0; r < 8; r++) inv[r] = 1.0f / lrun[r];
        float* dst = att + (size_t)di * 256 * 1024;
        #pragma unroll
        for (int s = 0; s < 16; s++) {
            const int col = h * 256 + s * 16 + n;
            #pragma unroll
            for (int r = 0; r < 8; r++) {
                const int row = qrow0 + g * 8 + r;
                dst[(size_t)row * 1024 + col] = oacc[s][r] * inv[r];
            }
        }
    }
}

// =====================================================================
// combined = sum_i softmax(dil_w)[i] * att[i]
// =====================================================================
__global__ __launch_bounds__(256) void combine_kernel(
    const float* __restrict__ att, const float* __restrict__ dilw,
    float* __restrict__ comb)
{
    const int i = blockIdx.x * blockDim.x + threadIdx.x;
    const float w0 = dilw[0], w1 = dilw[1], w2 = dilw[2], w3 = dilw[3];
    const float mx = fmaxf(fmaxf(w0, w1), fmaxf(w2, w3));
    const float e0 = __expf(w0 - mx), e1 = __expf(w1 - mx);
    const float e2 = __expf(w2 - mx), e3 = __expf(w3 - mx);
    const float inv = 1.0f / (e0 + e1 + e2 + e3);
    const int S = 256 * 1024;
    comb[i] = inv * (e0 * att[i] + e1 * att[S + i] + e2 * att[2 * S + i] + e3 * att[3 * S + i]);
}

// =====================================================================
// LayerNorm per row of 1024
// =====================================================================
__global__ __launch_bounds__(256) void ln_kernel(
    const float* __restrict__ y, const float* __restrict__ gamma,
    const float* __restrict__ beta, float* __restrict__ out)
{
    const int row = blockIdx.x;
    const float* yr = y + (size_t)row * 1024;
    float s = 0.0f, s2 = 0.0f;
    for (int i = threadIdx.x; i < 1024; i += 256) {
        float v = yr[i];
        s += v; s2 += v * v;
    }
    #pragma unroll
    for (int m = 1; m < 32; m <<= 1) { s += __shfl_xor(s, m); s2 += __shfl_xor(s2, m); }
    __shared__ float sh[16];
    const int wave = threadIdx.x >> 5, lane = threadIdx.x & 31;
    if (lane == 0) { sh[wave] = s; sh[8 + wave] = s2; }
    __syncthreads();
    if (wave == 0) {
        float a = (lane < 8) ? sh[lane] : 0.0f;
        float b = (lane < 8) ? sh[8 + lane] : 0.0f;
        #pragma unroll
        for (int m = 1; m < 8; m <<= 1) { a += __shfl_xor(a, m); b += __shfl_xor(b, m); }
        if (lane == 0) { sh[0] = a; sh[1] = b; }
    }
    __syncthreads();
    const float mean = sh[0] * (1.0f / 1024.0f);
    const float var  = sh[1] * (1.0f / 1024.0f) - mean * mean;
    const float rstd = rsqrtf(var + 1e-5f);
    for (int i = threadIdx.x; i < 1024; i += 256)
        out[(size_t)row * 1024 + i] = (yr[i] - mean) * rstd * gamma[i] + beta[i];
}

// =====================================================================
extern "C" void kernel_launch(void* const* d_in, const int* in_sizes, int n_in,
                              void* d_out, int out_size, void* d_ws, size_t ws_size,
                              hipStream_t stream)
{
    const float* x    = (const float*)d_in[0];   // (256,1024)
    const float* hist = (const float*)d_in[1];   // (64,256,1024) -> rows l*256+b
    const float* Wq   = (const float*)d_in[2];
    const float* bq   = (const float*)d_in[3];
    const float* Wk   = (const float*)d_in[4];
    const float* bk   = (const float*)d_in[5];
    const float* Wv   = (const float*)d_in[6];
    const float* bv   = (const float*)d_in[7];
    const float* Wo   = (const float*)d_in[8];
    const float* bo   = (const float*)d_in[9];
    const float* dilw = (const float*)d_in[10];
    const float* gamma= (const float*)d_in[11];
    const float* beta = (const float*)d_in[12];
    float* out = (float*)d_out;

    char* w = (char*)d_ws;
    unsigned short* Qb = (unsigned short*)w;                                  // 0.5 MB
    unsigned short* Kb = (unsigned short*)(w + (size_t)(1u << 19));           // 32 MB
    unsigned short* Vb = (unsigned short*)(w + (size_t)(1u << 19) + (1u << 25));
    float* att  = (float*)(w + (size_t)(1u << 19) + ((size_t)1u << 26));      // 4 MB
    float* comb = att + 4 * 256 * 1024;                                       // 1 MB
    float* ybuf = comb + 256 * 1024;                                          // 1 MB

    // Q / K / V projections (K,V once over ALL 64 frames — dilations reuse subsets)
    gemm_bias_kernel<true, false><<<dim3(4, 16),   128, 0, stream>>>(x,    Wq, bq, nullptr, Qb, 256,   1024, 1024);
    gemm_bias_kernel<true, false><<<dim3(256, 16), 128, 0, stream>>>(hist, Wk, bk, nullptr, Kb, 16384, 1024, 1024);
    gemm_bias_kernel<true, false><<<dim3(256, 16), 128, 0, stream>>>(hist, Wv, bv, nullptr, Vb, 16384, 1024, 1024);

    // Flash attention: 4 q-tiles x 4 heads x 4 dilations
    attn_kernel<<<dim3(4, 4, 4), 128, 0, stream>>>(Qb, Kb, Vb, att);

    // Weighted combine over dilations
    combine_kernel<<<dim3(1024), 256, 0, stream>>>(att, dilw, comb);

    // y = x + combined @ Wo + bo
    gemm_bias_kernel<false, true><<<dim3(4, 16), 128, 0, stream>>>(comb, Wo, bo, x, ybuf, 256, 1024, 1024);

    // LayerNorm -> output
    ln_kernel<<<dim3(256), 256, 0, stream>>>(ybuf, gamma, beta, out);
}
